// NarrowAttention_84224308674917
// MI455X (gfx1250) — compile-verified, matrix operand order checked by reading, not run on hardware
//
#include <hip/hip_runtime.h>
#include <hip/hip_bf16.h>

// ---------------------------------------------------------------------------
// NarrowAttention for MI455X (gfx1250): f16 WMMA pipeline, f32 accumulation.
//   1) convert f32 -> f16 (activations + weights)
//   2) NT GEMM (WMMA f32_16x16x32_f16, register double-buffered K loop):
//      Q,K -> [B,H,S,64], V -> [B,H,64,S] (transposed)
//   3) flash attention: block-level LDS staging of K/V chunks via
//      GLOBAL_LOAD_ASYNC_TO_LDS_B128 (ASYNCcnt), per-wave 16-row q tiles,
//      WMMA scores + WMMA P*V
//   4) NT GEMM output projection -> f32 d_out
// ---------------------------------------------------------------------------

typedef __attribute__((ext_vector_type(16))) _Float16 v16h;
typedef __attribute__((ext_vector_type(8)))  _Float16 v8h;
typedef __attribute__((ext_vector_type(4)))  _Float16 v4h;
typedef __attribute__((ext_vector_type(8)))  float    v8f;
typedef __attribute__((ext_vector_type(4)))  float    v4f;
typedef __attribute__((ext_vector_type(4)))  int      v4i;

#define D_MODEL 1024
#define N_HEADS 16
#define DEPTH   64
#define SEQ     2048
#define BATCH   2
#define M_TOTAL (BATCH * SEQ)   // 4096

// ---- CDNA5 async global->LDS path (guarded; fallback = load + ds_store) ----
#ifndef HAVE_ASYNC_LDS
#if defined(__has_builtin)
#if __has_builtin(__builtin_amdgcn_global_load_async_to_lds_b128)
#define HAVE_ASYNC_LDS 1
#endif
#endif
#endif
#ifndef HAVE_ASYNC_LDS
#define HAVE_ASYNC_LDS 0
#endif

#if HAVE_ASYNC_LDS
typedef __attribute__((address_space(1))) v4i gv4i;   // global int4
typedef __attribute__((address_space(3))) v4i lv4i;   // LDS int4
#endif

static __device__ __forceinline__ void async_copy_b128(const _Float16* g,
                                                       _Float16* l) {
#if HAVE_ASYNC_LDS
    __builtin_amdgcn_global_load_async_to_lds_b128(
        (gv4i*)g, (lv4i*)l, /*offset=*/0, /*cpol=*/0);
#else
    *(v8h*)l = *(const v8h*)g;
#endif
}

static __device__ __forceinline__ void wait_async0() {
#if HAVE_ASYNC_LDS
#if defined(__has_builtin) && __has_builtin(__builtin_amdgcn_s_wait_asynccnt)
    __builtin_amdgcn_s_wait_asynccnt(0);
#else
    asm volatile("s_wait_asynccnt 0x0" ::: "memory");
#endif
#endif
}

union Frag { v16h v; v8h h[2]; };

static __device__ __forceinline__ v8h ldg8h(const _Float16* p) {
    return *(const v8h*)p;
}

static __device__ __forceinline__ v8f wmma_f16(const Frag& a, const Frag& b, v8f c) {
    return __builtin_amdgcn_wmma_f32_16x16x32_f16(
        /*neg_a=*/false, a.v, /*neg_b=*/false, b.v,
        /*c_mod=*/(short)0, c, /*reuse_a=*/false, /*reuse_b=*/false);
}

// ---------------------------------------------------------------------------
// f32 -> f16 elementwise convert (4 elements / thread)
// ---------------------------------------------------------------------------
__global__ __launch_bounds__(256)
void cvt_f32_to_f16(const float* __restrict__ in, _Float16* __restrict__ out, int n4) {
    int i = blockIdx.x * blockDim.x + threadIdx.x;
    if (i < n4) {
        v4f x = *(const v4f*)(in + (size_t)i * 4);
        v4h y;
        y[0] = (_Float16)x[0]; y[1] = (_Float16)x[1];
        y[2] = (_Float16)x[2]; y[3] = (_Float16)x[3];
        *(v4h*)(out + (size_t)i * 4) = y;
    }
}

// ---------------------------------------------------------------------------
// NT GEMM: C[M,N] = A[M,K] @ Wt[N,K]^T + bias[N]   (K % 64 == 0)
// Block: 256 threads (8 waves). Wave tile 16(M) x 64(N); block tile 64x128.
// Register double-buffered K loop: load chunk k+32 while WMMAing chunk k.
// mode 0: f16 out, scatter to [B,H,S,DEPTH]       (Q, K)
// mode 1: f16 out, scatter to [B,H,DEPTH,S]       (V transposed)
// mode 2: f32 out, row-major [M,N]                (final projection)
// ---------------------------------------------------------------------------
__global__ __launch_bounds__(256)
void gemm_nt_f16(const _Float16* __restrict__ A,
                 const _Float16* __restrict__ Wt,
                 const float*    __restrict__ bias,
                 _Float16* __restrict__ outH,
                 float*    __restrict__ outF,
                 int M, int N, int K, int mode)
{
    const int lane = threadIdx.x & 31;
    const int w    = threadIdx.x >> 5;
    const int wm   = w & 3;
    const int wn   = w >> 2;
    const int row0 = blockIdx.x * 64 + wm * 16;
    const int col0 = blockIdx.y * 128 + wn * 64;
    const int r    = lane & 15;
    const int hi   = (lane >> 4) & 1;
    const int koff = hi * 8;   // ISA 16-bit A/B fragment: two 8-half chunks

    v8f zero = {0.f, 0.f, 0.f, 0.f, 0.f, 0.f, 0.f, 0.f};
    v8f acc[4] = {zero, zero, zero, zero};

    const _Float16* arow = A + (size_t)(row0 + r) * K;
    const _Float16* brow[4];
#pragma unroll
    for (int nt = 0; nt < 4; ++nt)
        brow[nt] = Wt + (size_t)(col0 + nt * 16 + r) * K;

    auto load_tile = [&](int k0, Frag& a, Frag (&b)[4]) {
        a.h[0] = ldg8h(arow + k0 + koff);
        a.h[1] = ldg8h(arow + k0 + koff + 16);
#pragma unroll
        for (int nt = 0; nt < 4; ++nt) {
            b[nt].h[0] = ldg8h(brow[nt] + k0 + koff);
            b[nt].h[1] = ldg8h(brow[nt] + k0 + koff + 16);
        }
    };

    Frag a0, b0[4], a1, b1[4];
    load_tile(0, a0, b0);

    for (int k0 = 0; k0 < K; k0 += 64) {
        __builtin_prefetch(arow + k0 + 128, 0, 1);   // global_prefetch_b8
        load_tile(k0 + 32, a1, b1);                  // in flight during WMMAs
#pragma unroll
        for (int nt = 0; nt < 4; ++nt) acc[nt] = wmma_f16(a0, b0[nt], acc[nt]);
        if (k0 + 64 < K) load_tile(k0 + 64, a0, b0);
#pragma unroll
        for (int nt = 0; nt < 4; ++nt) acc[nt] = wmma_f16(a1, b1[nt], acc[nt]);
    }

    // C layout: element e of v8f -> row (row0 + e + 8*hi), col (col0+nt*16+(lane&15))
    const int rowe = row0 + hi * 8;
#pragma unroll
    for (int e = 0; e < 8; ++e) {
        const int row = rowe + e;
#pragma unroll
        for (int nt = 0; nt < 4; ++nt) {
            const int col = col0 + nt * 16 + (lane & 15);
            const float val = acc[nt][e] + bias[col];
            if (mode == 2) {
                outF[(size_t)row * N + col] = val;
            } else {
                const int b = row >> 11;       // / SEQ
                const int s = row & (SEQ - 1);
                const int h = col >> 6;        // / DEPTH
                const int d = col & (DEPTH - 1);
                if (mode == 0)
                    outH[(((size_t)(b * N_HEADS + h) * SEQ) + s) * DEPTH + d] =
                        (_Float16)val;
                else
                    outH[(((size_t)(b * N_HEADS + h) * DEPTH) + d) * SEQ + s] =
                        (_Float16)val;
            }
        }
    }
}

// ---------------------------------------------------------------------------
// Flash attention. Q,K: [B*H, S, 64] f16.  Vt: [B*H, 64, S] f16.
// ctx out: [B, S, H*64] f16 (heads merged -> ready for output projection).
// Block = 8 waves = 128 q rows; grid (B*H, S/128).
// Per 32-key chunk: block cooperatively stages K (32x64) and V^T (64x32)
// tiles into double-buffered LDS (async-to-LDS when available), then each
// wave runs 4 score WMMAs + online softmax + 4 P*V WMMAs from LDS.
// ---------------------------------------------------------------------------
__global__ __launch_bounds__(256)
void flash_attn(const _Float16* __restrict__ Q,
                const _Float16* __restrict__ Kc,
                const _Float16* __restrict__ Vt,
                _Float16* __restrict__ ctx)
{
    __shared__ __align__(16) _Float16 Kbuf[2][32][DEPTH];  // 8 KB
    __shared__ __align__(16) _Float16 Vbuf[2][DEPTH][32];  // 8 KB
    __shared__ __align__(16) _Float16 ps[8][16 * 32];      // 8 KB (per-wave P)

    const int tid  = threadIdx.x;
    const int lane = tid & 31;
    const int w    = tid >> 5;
    const int r    = lane & 15;
    const int hi   = (lane >> 4) & 1;
    const int koff = hi * 8;
    const int bh   = blockIdx.x;
    const int q0   = blockIdx.y * 128 + w * 16;

    const _Float16* Qp = Q  + (size_t)bh * SEQ * DEPTH;
    const _Float16* Kp = Kc + (size_t)bh * SEQ * DEPTH;
    const _Float16* Vp = Vt + (size_t)bh * DEPTH * SEQ;

    // Cooperative staging of chunk c (keys [32c, 32c+32)) into buffer c&1.
    // K tile: thread t -> row t>>3 (0..31), 8 halves at col (t&7)*8.
    // V tile: thread t -> row t>>2 (0..63), 8 halves at col (t&3)*8.
    auto issue_chunk = [&](int c) {
        const int buf = c & 1;
        const int kc  = c * 32;
        async_copy_b128(Kp + (size_t)(kc + (tid >> 3)) * DEPTH + (tid & 7) * 8,
                        &Kbuf[buf][tid >> 3][(tid & 7) * 8]);
        async_copy_b128(Vp + (size_t)(tid >> 2) * SEQ + kc + (tid & 3) * 8,
                        &Vbuf[buf][tid >> 2][(tid & 3) * 8]);
    };

    // Q fragments: rows q0..q0+15, depth split into K=0..31 and K=32..63
    Frag aq0, aq1;
    {
        const _Float16* qp = Qp + (size_t)(q0 + r) * DEPTH;
        aq0.h[0] = ldg8h(qp + koff);        aq0.h[1] = ldg8h(qp + koff + 16);
        aq1.h[0] = ldg8h(qp + 32 + koff);   aq1.h[1] = ldg8h(qp + 32 + koff + 16);
    }

    v8f zero = {0.f, 0.f, 0.f, 0.f, 0.f, 0.f, 0.f, 0.f};
    v8f accd[4] = {zero, zero, zero, zero};
    float mrun[8], lrun[8];
#pragma unroll
    for (int e = 0; e < 8; ++e) { mrun[e] = -3.402823466e38f; lrun[e] = 0.f; }

    const float scale = 0.125f;  // 1/sqrt(64)
    const int NC = SEQ / 32;     // 64 chunks

    issue_chunk(0);

    for (int c = 0; c < NC; ++c) {
        wait_async0();        // this wave's async copies for chunk c complete
        __syncthreads();      // chunk c visible; everyone done with chunk c-1
        if (c + 1 < NC) issue_chunk(c + 1);   // fill the other buffer

        const int buf = c & 1;

        // ---- scores: two 16x16 tiles covering keys [32c, 32c+32) ----
        v8f sc[2];
#pragma unroll
        for (int nt = 0; nt < 2; ++nt) {
            const _Float16* kp = &Kbuf[buf][nt * 16 + r][0];
            Frag f0, f1;
            f0.h[0] = *(const v8h*)(kp + koff);
            f0.h[1] = *(const v8h*)(kp + koff + 16);
            f1.h[0] = *(const v8h*)(kp + 32 + koff);
            f1.h[1] = *(const v8h*)(kp + 32 + koff + 16);
            v8f cacc = zero;
            cacc = wmma_f16(aq0, f0, cacc);
            cacc = wmma_f16(aq1, f1, cacc);
            sc[nt] = cacc;
        }

        // ---- online softmax (row lives across 16 lanes of a half-wave) ----
#pragma unroll
        for (int e = 0; e < 8; ++e) {
            float s0 = sc[0][e] * scale;
            float s1 = sc[1][e] * scale;
            float t  = fmaxf(s0, s1);
            t = fmaxf(t, __shfl_xor(t, 8, 32));
            t = fmaxf(t, __shfl_xor(t, 4, 32));
            t = fmaxf(t, __shfl_xor(t, 2, 32));
            t = fmaxf(t, __shfl_xor(t, 1, 32));
            const float mnew  = fmaxf(mrun[e], t);
            const float alpha = __expf(mrun[e] - mnew);
            const float p0 = __expf(s0 - mnew);
            const float p1 = __expf(s1 - mnew);
            float rs = p0 + p1;
            rs += __shfl_xor(rs, 8, 32);
            rs += __shfl_xor(rs, 4, 32);
            rs += __shfl_xor(rs, 2, 32);
            rs += __shfl_xor(rs, 1, 32);
            lrun[e] = lrun[e] * alpha + rs;
            mrun[e] = mnew;
#pragma unroll
            for (int dt = 0; dt < 4; ++dt) accd[dt][e] *= alpha;
            const int prow = e + hi * 8;
            ps[w][prow * 32 + (lane & 15)]      = (_Float16)p0;
            ps[w][prow * 32 + 16 + (lane & 15)] = (_Float16)p1;
        }

        // ---- re-swizzle P from C-layout to A-fragment via LDS ----
        Frag pa;
        pa.h[0] = *(const v8h*)&ps[w][r * 32 + koff];
        pa.h[1] = *(const v8h*)&ps[w][r * 32 + koff + 16];

        // ---- ctx += P @ V^T  (Vbuf rows are depth, K-dim is keys) ----
#pragma unroll
        for (int dt = 0; dt < 4; ++dt) {
            const _Float16* vp = &Vbuf[buf][dt * 16 + r][0];
            Frag bv;
            bv.h[0] = *(const v8h*)(vp + koff);
            bv.h[1] = *(const v8h*)(vp + koff + 16);
            accd[dt] = wmma_f16(pa, bv, accd[dt]);
        }
    }

    // ---- normalize and write merged-head ctx [B, S, D_MODEL] ----
    const int b = bh >> 4;
    const int h = bh & 15;
#pragma unroll
    for (int e = 0; e < 8; ++e) {
        const float inv = 1.0f / lrun[e];
        const int srow  = q0 + e + hi * 8;
#pragma unroll
        for (int dt = 0; dt < 4; ++dt) {
            const int col = h * DEPTH + dt * 16 + (lane & 15);
            ctx[((size_t)(b * SEQ + srow)) * D_MODEL + col] =
                (_Float16)(accd[dt][e] * inv);
        }
    }
}

// ---------------------------------------------------------------------------
extern "C" void kernel_launch(void* const* d_in, const int* in_sizes, int n_in,
                              void* d_out, int out_size, void* d_ws, size_t ws_size,
                              hipStream_t stream) {
    (void)in_sizes; (void)n_in; (void)out_size; (void)ws_size;

    const float* q_in = (const float*)d_in[0];
    const float* k_in = (const float*)d_in[1];
    const float* v_in = (const float*)d_in[2];
    const float* Wq   = (const float*)d_in[3];
    const float* bq   = (const float*)d_in[4];
    const float* Wk   = (const float*)d_in[5];
    const float* bk   = (const float*)d_in[6];
    const float* Wv   = (const float*)d_in[7];
    const float* bv   = (const float*)d_in[8];
    const float* Wo   = (const float*)d_in[9];
    const float* bo   = (const float*)d_in[10];

    char* ws = (char*)d_ws;
    size_t off = 0;
    auto alloc = [&](size_t bytes) -> void* {
        void* p = ws + off;
        off += (bytes + 255) & ~(size_t)255;
        return p;
    };

    const size_t actB = (size_t)M_TOTAL * D_MODEL * sizeof(_Float16);   // 8 MiB
    const size_t wB   = (size_t)D_MODEL * D_MODEL * sizeof(_Float16);   // 2 MiB

    _Float16* Xq  = (_Float16*)alloc(actB);
    _Float16* Xk  = (_Float16*)alloc(actB);
    _Float16* Xv  = (_Float16*)alloc(actB);
    _Float16* Whq = (_Float16*)alloc(wB);
    _Float16* Whk = (_Float16*)alloc(wB);
    _Float16* Whv = (_Float16*)alloc(wB);
    _Float16* Who = (_Float16*)alloc(wB);
    _Float16* Qh  = (_Float16*)alloc(actB);
    _Float16* Kh  = (_Float16*)alloc(actB);
    _Float16* Vt  = (_Float16*)alloc(actB);
    _Float16* Ctx = (_Float16*)alloc(actB);

    // 1) convert to f16
    const int nAct4 = M_TOTAL * D_MODEL / 4;      // 1,048,576
    const int nW4   = D_MODEL * D_MODEL / 4;      //   262,144
    cvt_f32_to_f16<<<nAct4 / 256, 256, 0, stream>>>(q_in, Xq, nAct4);
    cvt_f32_to_f16<<<nAct4 / 256, 256, 0, stream>>>(k_in, Xk, nAct4);
    cvt_f32_to_f16<<<nAct4 / 256, 256, 0, stream>>>(v_in, Xv, nAct4);
    cvt_f32_to_f16<<<nW4 / 256, 256, 0, stream>>>(Wq, Whq, nW4);
    cvt_f32_to_f16<<<nW4 / 256, 256, 0, stream>>>(Wk, Whk, nW4);
    cvt_f32_to_f16<<<nW4 / 256, 256, 0, stream>>>(Wv, Whv, nW4);
    cvt_f32_to_f16<<<nW4 / 256, 256, 0, stream>>>(Wo, Who, nW4);

    // 2) QKV projections (NT GEMM + scatter epilogues)
    dim3 ggrid(M_TOTAL / 64, D_MODEL / 128);
    gemm_nt_f16<<<ggrid, 256, 0, stream>>>(Xq, Whq, bq, Qh, nullptr,
                                           M_TOTAL, D_MODEL, D_MODEL, 0);
    gemm_nt_f16<<<ggrid, 256, 0, stream>>>(Xk, Whk, bk, Kh, nullptr,
                                           M_TOTAL, D_MODEL, D_MODEL, 0);
    gemm_nt_f16<<<ggrid, 256, 0, stream>>>(Xv, Whv, bv, Vt, nullptr,
                                           M_TOTAL, D_MODEL, D_MODEL, 1);

    // 3) flash attention
    dim3 agrid(BATCH * N_HEADS, SEQ / 128);
    flash_attn<<<agrid, 256, 0, stream>>>(Qh, Kh, Vt, Ctx);

    // 4) output projection -> f32 d_out
    gemm_nt_f16<<<ggrid, 256, 0, stream>>>(Ctx, Who, bo, nullptr, (float*)d_out,
                                           M_TOTAL, D_MODEL, D_MODEL, 2);
}
